// GATv2Encoder_47631187312716
// MI455X (gfx1250) — matrix operand order repaired and checked by reading, not compiled, not run on hardware
//
#include <hip/hip_runtime.h>

// ---------------------------------------------------------------------------
// Types for CDNA5 WMMA (wave32): 16x16x32 bf16 -> f32 accumulate
// ---------------------------------------------------------------------------
typedef __attribute__((ext_vector_type(16))) __bf16 v16bf;
typedef __attribute__((ext_vector_type(8)))  float  v8f;

union Frag { v16bf v; uint4 u[2]; };

// ---------------------------------------------------------------------------
// Elementwise helpers
// ---------------------------------------------------------------------------
__global__ void k_f32_to_bf16(const float* __restrict__ in, __bf16* __restrict__ out, size_t n) {
  size_t i = (size_t)blockIdx.x * blockDim.x + threadIdx.x;
  size_t stride = (size_t)gridDim.x * blockDim.x;
  for (; i < n; i += stride) out[i] = (__bf16)in[i];
}

__global__ void k_fill_f32(float* __restrict__ p, float v, size_t n) {
  size_t i = (size_t)blockIdx.x * blockDim.x + threadIdx.x;
  size_t stride = (size_t)gridDim.x * blockDim.x;
  for (; i < n; i += stride) p[i] = v;
}

// out_bf16[i] = bf16( act(in[i] + bias[i % C]) )
__global__ void k_bias_act_bf16(const float* __restrict__ in, const float* __restrict__ bias,
                                __bf16* __restrict__ out, size_t n, int C, int relu) {
  size_t i = (size_t)blockIdx.x * blockDim.x + threadIdx.x;
  size_t stride = (size_t)gridDim.x * blockDim.x;
  for (; i < n; i += stride) {
    float v = in[i] + bias[(int)(i % (size_t)C)];
    if (relu) v = fmaxf(v, 0.0f);
    out[i] = (__bf16)v;
  }
}

// ---------------------------------------------------------------------------
// Weight packing: W is [K x N] f32 row-major. Pack into WMMA B-fragment order:
// packed[((kt*tilesN + nt)*32 + lane)*16 + i] ; lanes 0-15 hold K=kbase..kbase+15
// for column (lane&15) of 16-col tile nt; lanes 16-31 hold K=kbase+16..+31.
// ---------------------------------------------------------------------------
__global__ void k_pack_weight(const float* __restrict__ W, __bf16* __restrict__ P, int K, int Nn) {
  const int tilesN = Nn >> 4;
  const int total = (K >> 5) * tilesN * 32;
  int idx = blockIdx.x * blockDim.x + threadIdx.x;
  if (idx >= total) return;
  const int lane = idx & 31;
  const int tile = idx >> 5;
  const int nt = tile % tilesN;
  const int kt = tile / tilesN;
  const int col = nt * 16 + (lane & 15);
  const int kbase = kt * 32 + ((lane & 16) ? 16 : 0);
  __bf16* dst = P + (size_t)idx * 16;
#pragma unroll
  for (int i = 0; i < 16; ++i)
    dst[i] = (__bf16)W[(size_t)(kbase + i) * Nn + col];
}

// ---------------------------------------------------------------------------
// WMMA GEMM, 2x2 register-blocked: each wave computes a 32(M) x 32(N) tile
// with 4 f32 accumulators; per K-step: 8 x b128 loads -> 4 x wmma (2:1).
// Requires N % 32 == 0, K % 32 == 0. M only needs M % 16 == 0: the second
// row-fragment clamps its loads to row M-1 and guards its stores.
// ---------------------------------------------------------------------------
__global__ void k_wmma_gemm(const __bf16* __restrict__ A, const __bf16* __restrict__ Bp,
                            const float* __restrict__ bias, float* __restrict__ C,
                            int M, int Nn, int K, int relu) {
  const int lane = threadIdx.x & 31;
  const int wave = threadIdx.x >> 5;
  const int tilesN2 = Nn >> 5;          // 32-col groups
  const int tilesM2 = (M + 31) >> 5;    // 32-row groups (last may be half)
  const int tile = blockIdx.x * (blockDim.x >> 5) + wave;
  if (tile >= tilesM2 * tilesN2) return;
  const int tm = tile / tilesN2;
  const int tn = tile % tilesN2;
  const int half = lane >> 4;           // 0: K 0..15 ; 1: K 16..31
  const int lr = lane & 15;

  const int row0 = tm * 32 + lr;                       // always < M
  const int row1 = row0 + 16;
  const int row1c = (row1 < M) ? row1 : (M - 1);       // clamp loads on M-tail
  const __bf16* arow0 = A + (size_t)row0 * K;
  const __bf16* arow1 = A + (size_t)row1c * K;

  const int tilesN = Nn >> 4;
  const size_t bstride = (size_t)tilesN * 32 * 16;     // elements per K-tile slab
  const __bf16* bb0 = Bp + (((size_t)(tn * 2) * 32) + lane) * 16;
  const __bf16* bb1 = bb0 + 32 * 16;                   // next 16-col tile

  v8f acc00 = {}, acc01 = {}, acc10 = {}, acc11 = {};
  const int ksteps = K >> 5;
  for (int kt = 0; kt < ksteps; ++kt) {
    Frag a0, a1, b0, b1;
    const int ko = kt * 32 + half * 16;
    const __bf16* ap0 = arow0 + ko;
    const __bf16* ap1 = arow1 + ko;
    a0.u[0] = *(const uint4*)(const void*)(ap0);
    a0.u[1] = *(const uint4*)(const void*)(ap0 + 8);
    a1.u[0] = *(const uint4*)(const void*)(ap1);
    a1.u[1] = *(const uint4*)(const void*)(ap1 + 8);
    const __bf16* bp0 = bb0 + (size_t)kt * bstride;
    const __bf16* bp1 = bb1 + (size_t)kt * bstride;
    b0.u[0] = *(const uint4*)(const void*)(bp0);
    b0.u[1] = *(const uint4*)(const void*)(bp0 + 8);
    b1.u[0] = *(const uint4*)(const void*)(bp1);
    b1.u[1] = *(const uint4*)(const void*)(bp1 + 8);
    acc00 = __builtin_amdgcn_wmma_f32_16x16x32_bf16(false, a0.v, false, b0.v, (short)0, acc00, false, false);
    acc01 = __builtin_amdgcn_wmma_f32_16x16x32_bf16(false, a0.v, false, b1.v, (short)0, acc01, false, false);
    acc10 = __builtin_amdgcn_wmma_f32_16x16x32_bf16(false, a1.v, false, b0.v, (short)0, acc10, false, false);
    acc11 = __builtin_amdgcn_wmma_f32_16x16x32_bf16(false, a1.v, false, b1.v, (short)0, acc11, false, false);
  }

  // C/D layout: lanes 0-15 -> rows rbase..rbase+7 (half=0) / +8..15 (half=1); col = lane&15
  const int col0 = tn * 32 + lr;
  const int col1 = col0 + 16;
  const int rbase = tm * 32 + half * 8;
  const float bv0 = bias ? bias[col0] : 0.0f;
  const float bv1 = bias ? bias[col1] : 0.0f;
#pragma unroll
  for (int r = 0; r < 8; ++r) {
    float v0 = acc00[r] + bv0;
    float v1 = acc01[r] + bv1;
    if (relu) { v0 = fmaxf(v0, 0.0f); v1 = fmaxf(v1, 0.0f); }
    C[(size_t)(rbase + r) * Nn + col0] = v0;
    C[(size_t)(rbase + r) * Nn + col1] = v1;
  }
#pragma unroll
  for (int r = 0; r < 8; ++r) {
    const int row = rbase + 16 + r;
    if (row < M) {
      float v0 = acc10[r] + bv0;
      float v1 = acc11[r] + bv1;
      if (relu) { v0 = fmaxf(v0, 0.0f); v1 = fmaxf(v1, 0.0f); }
      C[(size_t)row * Nn + col0] = v0;
      C[(size_t)row * Nn + col1] = v1;
    }
  }
}

// ---------------------------------------------------------------------------
// Edge-phase kernels (GATv2 attention). Edges = E real edges + N self loops.
// ---------------------------------------------------------------------------
__device__ inline void atomicMaxF(float* addr, float val) {
  int* ia = (int*)addr;
  int old = __float_as_int(*addr);
  while (__int_as_float(old) < val) {
    int assumed = old;
    old = atomicCAS(ia, assumed, __float_as_int(val));
    if (old == assumed) break;
  }
}

// e[edge,h] = att[h,:] . leakyrelu(xl[src,h,:] + xr[dst,h,:]); atomicMax into emax[dst,h]
__global__ void k_edge_score(const float* __restrict__ xl, const float* __restrict__ xr,
                             const float* __restrict__ att, const int* __restrict__ ei,
                             float* __restrict__ e, float* __restrict__ emax,
                             int E, int Et, int H, int Cc) {
  int idx = blockIdx.x * blockDim.x + threadIdx.x;
  if (idx >= Et * H) return;
  const int edge = idx / H;
  const int h = idx - edge * H;
  const int src = (edge < E) ? ei[edge] : (edge - E);
  const int dst = (edge < E) ? ei[E + edge] : (edge - E);
  const float4* pl = (const float4*)(const void*)(xl + (size_t)src * H * Cc + h * Cc);
  const float4* pr = (const float4*)(const void*)(xr + (size_t)dst * H * Cc + h * Cc);
  const float4* pa = (const float4*)(const void*)(att + h * Cc);
  float s = 0.0f;
  const int c4 = Cc >> 2;
  for (int c = 0; c < c4; ++c) {
    float4 l = pl[c], r = pr[c], a = pa[c];
    float m0 = l.x + r.x, m1 = l.y + r.y, m2 = l.z + r.z, m3 = l.w + r.w;
    m0 = (m0 > 0.0f) ? m0 : 0.2f * m0;   // LeakyReLU(0.2)
    m1 = (m1 > 0.0f) ? m1 : 0.2f * m1;
    m2 = (m2 > 0.0f) ? m2 : 0.2f * m2;
    m3 = (m3 > 0.0f) ? m3 : 0.2f * m3;
    s += m0 * a.x + m1 * a.y + m2 * a.z + m3 * a.w;
  }
  e[idx] = s;
  atomicMaxF(&emax[dst * H + h], s);
}

// a = exp(e - emax[dst,h]); store; atomicAdd into denom[dst,h]
__global__ void k_edge_exp(float* __restrict__ e, const float* __restrict__ emax,
                           float* __restrict__ denom, const int* __restrict__ ei,
                           int E, int Et, int H) {
  int idx = blockIdx.x * blockDim.x + threadIdx.x;
  if (idx >= Et * H) return;
  const int edge = idx / H;
  const int h = idx - edge * H;
  const int dst = (edge < E) ? ei[E + edge] : (edge - E);
  float a = __expf(e[idx] - emax[dst * H + h]);
  e[idx] = a;
  atomicAdd(&denom[dst * H + h], a);
}

// One block per edge (blockDim = H*Cc). alpha computed once per head in LDS,
// broadcast to all channel threads: agg[dst,h,c] += alpha[h] * xl[src,h,c]
__global__ void k_edge_aggregate(const float* __restrict__ e, const float* __restrict__ denom,
                                 const float* __restrict__ xl, float* __restrict__ agg,
                                 const int* __restrict__ ei, int E, int H, int Cc) {
  const int edge = blockIdx.x;
  const int src = (edge < E) ? ei[edge] : (edge - E);
  const int dst = (edge < E) ? ei[E + edge] : (edge - E);
  __shared__ float salpha[8];
  if ((int)threadIdx.x < H)
    salpha[threadIdx.x] = e[edge * H + threadIdx.x] / denom[dst * H + threadIdx.x];
  __syncthreads();
  const int hc = threadIdx.x;                 // 0 .. H*Cc-1
  const float alpha = salpha[hc / Cc];
  atomicAdd(&agg[(size_t)dst * H * Cc + hc], alpha * xl[(size_t)src * H * Cc + hc]);
}

// ---------------------------------------------------------------------------
// BatchNorm (training-mode batch statistics over N rows, C channels)
// ---------------------------------------------------------------------------
__global__ void k_bn_reduce(const float* __restrict__ X, float* __restrict__ sum,
                            float* __restrict__ sumsq, int n, int Cc) {
  const int ch = threadIdx.x % Cc;
  const int sub = threadIdx.x / Cc;
  const int nsub = blockDim.x / Cc;
  float s = 0.0f, s2 = 0.0f;
  for (int row = blockIdx.x * nsub + sub; row < n; row += gridDim.x * nsub) {
    float v = X[(size_t)row * Cc + ch];
    s += v; s2 += v * v;
  }
  atomicAdd(&sum[ch], s);
  atomicAdd(&sumsq[ch], s2);
}

__global__ void k_bn_apply_bf16(const float* __restrict__ X, const float* __restrict__ sum,
                                const float* __restrict__ sumsq, const float* __restrict__ gamma,
                                const float* __restrict__ beta, __bf16* __restrict__ out,
                                int n, int Cc) {
  size_t idx = (size_t)blockIdx.x * blockDim.x + threadIdx.x;
  size_t stride = (size_t)gridDim.x * blockDim.x;
  const size_t total = (size_t)n * Cc;
  const float invn = 1.0f / (float)n;
  for (; idx < total; idx += stride) {
    const int ch = (int)(idx % (size_t)Cc);
    const float mu = sum[ch] * invn;
    const float var = sumsq[ch] * invn - mu * mu;
    float v = (X[idx] - mu) * rsqrtf(var + 1e-5f) * gamma[ch] + beta[ch];
    v = fmaxf(v, 0.0f);
    out[idx] = (__bf16)v;
  }
}

// ---------------------------------------------------------------------------
// Host orchestration
// ---------------------------------------------------------------------------
extern "C" void kernel_launch(void* const* d_in, const int* in_sizes, int n_in,
                              void* d_out, int out_size, void* d_ws, size_t ws_size,
                              hipStream_t stream) {
  (void)n_in; (void)in_sizes; (void)out_size; (void)ws_size;

  const int  Nn  = 50000;
  const int  Ee  = 800000;
  const int  Et  = Ee + Nn;       // + self loops
  const int  INC = 2048;
  const int  H1  = 4,  C1 = 64,  F1 = 256;   // layer1: 4 heads x 64, concat -> 256
  const int  F2  = 128;                      // layer2 / projection width

  const float* x    = (const float*)d_in[0];
  const int*   ei   = (const int*)d_in[1];
  const float* W1l  = (const float*)d_in[2];
  const float* W1r  = (const float*)d_in[3];
  const float* att1 = (const float*)d_in[4];
  const float* b1   = (const float*)d_in[5];
  const float* W2l  = (const float*)d_in[6];
  const float* W2r  = (const float*)d_in[7];
  const float* att2 = (const float*)d_in[8];
  const float* b2   = (const float*)d_in[9];
  const float* pw1  = (const float*)d_in[10];
  const float* pb1  = (const float*)d_in[11];
  const float* gam  = (const float*)d_in[12];
  const float* bet  = (const float*)d_in[13];
  const float* pw2  = (const float*)d_in[14];
  const float* pb2  = (const float*)d_in[15];
  float* out = (float*)d_out;

  // ---- workspace carving ----
  char* ws = (char*)d_ws;
  size_t off = 0;
  auto carve = [&](size_t bytes) -> char* {
    char* p = ws + off;
    off = (off + bytes + 255) & ~(size_t)255;
    return p;
  };
  // Region 0: x in bf16 (only live until the first two GEMMs finish) — reused after.
  char*   xbf_raw = carve((size_t)Nn * INC * 2);
  float*  xl1    = (float*)carve((size_t)Nn * F1 * 4);
  float*  xr1    = (float*)carve((size_t)Nn * F1 * 4);
  float*  agg1   = (float*)carve((size_t)Nn * F1 * 4);
  float*  e1     = (float*)carve((size_t)Et * H1 * 4);
  float*  emax1  = (float*)carve((size_t)Nn * H1 * 4);
  float*  den1   = (float*)carve((size_t)Nn * H1 * 4);
  __bf16* w1lp   = (__bf16*)carve((size_t)(INC/32) * (F1/16) * 32 * 16 * 2);
  __bf16* w1rp   = (__bf16*)carve((size_t)(INC/32) * (F1/16) * 32 * 16 * 2);
  __bf16* w2lp   = (__bf16*)carve((size_t)(F1/32) * (F2/16) * 32 * 16 * 2);
  __bf16* w2rp   = (__bf16*)carve((size_t)(F1/32) * (F2/16) * 32 * 16 * 2);
  __bf16* pw1p   = (__bf16*)carve((size_t)(F2/32) * (F2/16) * 32 * 16 * 2);
  __bf16* pw2p   = (__bf16*)carve((size_t)(F2/32) * (F2/16) * 32 * 16 * 2);

  // Reuse region inside xbf (everything below is only used after xbf is dead).
  size_t roff = 0;
  auto reuse = [&](size_t bytes) -> char* {
    char* p = xbf_raw + roff;
    roff = (roff + bytes + 255) & ~(size_t)255;
    return p;
  };
  __bf16* xbf   = (__bf16*)xbf_raw;
  __bf16* h1bf  = (__bf16*)reuse((size_t)Nn * F1 * 2);
  float*  xl2   = (float*)reuse((size_t)Nn * F2 * 4);
  float*  xr2   = (float*)reuse((size_t)Nn * F2 * 4);
  float*  agg2  = (float*)reuse((size_t)Nn * F2 * 4);
  float*  e2    = (float*)reuse((size_t)Et * 4);
  float*  emax2 = (float*)reuse((size_t)Nn * 4);
  float*  den2  = (float*)reuse((size_t)Nn * 4);
  __bf16* h2bf  = (__bf16*)reuse((size_t)Nn * F2 * 2);
  float*  p1    = (float*)reuse((size_t)Nn * F2 * 4);
  __bf16* p1nbf = (__bf16*)reuse((size_t)Nn * F2 * 2);
  float*  bnsum = (float*)reuse(F2 * 4);
  float*  bnsq  = (float*)reuse(F2 * 4);

  auto nblk = [](long long n, int t) { return (unsigned)((n + t - 1) / t); };
  auto gemm_blocks = [&](int M, int Nc) {
    long long tiles = (long long)((M + 31) / 32) * (Nc / 32);
    return (unsigned)((tiles + 3) / 4);     // 4 waves (128 threads) per block
  };

  // ===== Layer 1: GEMMs (bf16 WMMA) =====
  k_f32_to_bf16<<<4096, 256, 0, stream>>>(x, xbf, (size_t)Nn * INC);
  k_pack_weight<<<nblk((INC/32)*(F1/16)*32, 256), 256, 0, stream>>>(W1l, w1lp, INC, F1);
  k_pack_weight<<<nblk((INC/32)*(F1/16)*32, 256), 256, 0, stream>>>(W1r, w1rp, INC, F1);
  k_wmma_gemm<<<gemm_blocks(Nn, F1), 128, 0, stream>>>(xbf, w1lp, nullptr, xl1, Nn, F1, INC, 0);
  k_wmma_gemm<<<gemm_blocks(Nn, F1), 128, 0, stream>>>(xbf, w1rp, nullptr, xr1, Nn, F1, INC, 0);

  // ===== Layer 1: edge softmax + aggregation =====
  k_fill_f32<<<nblk((long long)Nn * H1, 256), 256, 0, stream>>>(emax1, -3.0e38f, (size_t)Nn * H1);
  k_fill_f32<<<nblk((long long)Nn * H1, 256), 256, 0, stream>>>(den1, 0.0f, (size_t)Nn * H1);
  k_fill_f32<<<nblk((long long)Nn * F1, 256), 256, 0, stream>>>(agg1, 0.0f, (size_t)Nn * F1);
  k_edge_score<<<nblk((long long)Et * H1, 256), 256, 0, stream>>>(xl1, xr1, att1, ei, e1, emax1, Ee, Et, H1, C1);
  k_edge_exp<<<nblk((long long)Et * H1, 256), 256, 0, stream>>>(e1, emax1, den1, ei, Ee, Et, H1);
  k_edge_aggregate<<<Et, H1 * C1, 0, stream>>>(e1, den1, xl1, agg1, ei, Ee, H1, C1);
  k_bias_act_bf16<<<4096, 256, 0, stream>>>(agg1, b1, h1bf, (size_t)Nn * F1, F1, 1);

  // ===== Layer 2: GEMMs =====
  k_pack_weight<<<nblk((F1/32)*(F2/16)*32, 256), 256, 0, stream>>>(W2l, w2lp, F1, F2);
  k_pack_weight<<<nblk((F1/32)*(F2/16)*32, 256), 256, 0, stream>>>(W2r, w2rp, F1, F2);
  k_wmma_gemm<<<gemm_blocks(Nn, F2), 128, 0, stream>>>(h1bf, w2lp, nullptr, xl2, Nn, F2, F1, 0);
  k_wmma_gemm<<<gemm_blocks(Nn, F2), 128, 0, stream>>>(h1bf, w2rp, nullptr, xr2, Nn, F2, F1, 0);

  // ===== Layer 2: edge softmax + aggregation (H=1, C=128; mean over 1 head = identity) =====
  k_fill_f32<<<nblk(Nn, 256), 256, 0, stream>>>(emax2, -3.0e38f, (size_t)Nn);
  k_fill_f32<<<nblk(Nn, 256), 256, 0, stream>>>(den2, 0.0f, (size_t)Nn);
  k_fill_f32<<<nblk((long long)Nn * F2, 256), 256, 0, stream>>>(agg2, 0.0f, (size_t)Nn * F2);
  k_edge_score<<<nblk(Et, 256), 256, 0, stream>>>(xl2, xr2, att2, ei, e2, emax2, Ee, Et, 1, F2);
  k_edge_exp<<<nblk(Et, 256), 256, 0, stream>>>(e2, emax2, den2, ei, Ee, Et, 1);
  k_edge_aggregate<<<Et, F2, 0, stream>>>(e2, den2, xl2, agg2, ei, Ee, 1, F2);
  k_bias_act_bf16<<<4096, 256, 0, stream>>>(agg2, b2, h2bf, (size_t)Nn * F2, F2, 0);

  // ===== Projection head: Linear -> BN -> ReLU -> Linear =====
  k_pack_weight<<<nblk((F2/32)*(F2/16)*32, 256), 256, 0, stream>>>(pw1, pw1p, F2, F2);
  k_pack_weight<<<nblk((F2/32)*(F2/16)*32, 256), 256, 0, stream>>>(pw2, pw2p, F2, F2);
  k_wmma_gemm<<<gemm_blocks(Nn, F2), 128, 0, stream>>>(h2bf, pw1p, pb1, p1, Nn, F2, F2, 0);
  k_fill_f32<<<1, 256, 0, stream>>>(bnsum, 0.0f, (size_t)F2);
  k_fill_f32<<<1, 256, 0, stream>>>(bnsq, 0.0f, (size_t)F2);
  k_bn_reduce<<<256, 256, 0, stream>>>(p1, bnsum, bnsq, Nn, F2);
  k_bn_apply_bf16<<<4096, 256, 0, stream>>>(p1, bnsum, bnsq, gam, bet, p1nbf, Nn, F2);
  k_wmma_gemm<<<gemm_blocks(Nn, F2), 128, 0, stream>>>(p1nbf, pw2p, pb2, out, Nn, F2, F2, 0);
}
